// GIN_12292196401224
// MI455X (gfx1250) — compile-verified, matrix operand order checked by reading, not run on hardware
//
#include <hip/hip_runtime.h>
#include <stddef.h>

#define HDIM 128

typedef __attribute__((ext_vector_type(16))) __bf16        v16bf;
typedef __attribute__((ext_vector_type(8)))  float         v8f;
typedef __attribute__((ext_vector_type(4)))  unsigned int  v4u;

union Frag {
    v16bf v;
    struct { v4u lo; v4u hi; } u;
};

__device__ __forceinline__ unsigned short f2bf(float f) {
    unsigned int u = __float_as_uint(f);
    unsigned int r = u + 0x7FFFu + ((u >> 16) & 1u);   // round-to-nearest-even
    return (unsigned short)(r >> 16);
}
__device__ __forceinline__ float bf2f(unsigned short h) {
    return __uint_as_float(((unsigned int)h) << 16);
}

// ---------------------------------------------------------------------------
// Per-layer weight prep: Wt[n][k] = bf16(Wa[k][n]) (zero-padded K), plus
// folded BN epilogue constants:  out = relu(acc*c1 + c0)
// ---------------------------------------------------------------------------
__global__ void gin_prep_weights(const float* __restrict__ Wa, int din, int KPAD,
                                 const float* __restrict__ Wb,
                                 const float* __restrict__ ba, const float* __restrict__ g,
                                 const float* __restrict__ be, const float* __restrict__ rm,
                                 const float* __restrict__ rv, const float* __restrict__ bb,
                                 unsigned short* __restrict__ WtA,
                                 unsigned short* __restrict__ WtB,
                                 float* __restrict__ c0a, float* __restrict__ c1a,
                                 float* __restrict__ c0b, float* __restrict__ c1b)
{
    int idx = blockIdx.x * blockDim.x + threadIdx.x;
    if (idx < HDIM * KPAD) {
        int n = idx / KPAD, k = idx % KPAD;
        float v = (k < din) ? Wa[(size_t)k * HDIM + n] : 0.f;
        WtA[idx] = f2bf(v);
    }
    if (idx < HDIM * HDIM) {
        int n = idx / HDIM, k = idx % HDIM;
        WtB[idx] = f2bf(Wb[(size_t)k * HDIM + n]);
    }
    if (idx < HDIM) {
        float s = g[idx] * rsqrtf(rv[idx] + 1e-5f);
        c1a[idx] = s;
        c0a[idx] = (ba[idx] - rm[idx]) * s + be[idx];
        c1b[idx] = 1.f;
        c0b[idx] = bb[idx];
    }
}

// ---------------------------------------------------------------------------
// fA[i][k] = (k<K) ? input(i,k) : 0      (input = x fp32 or hB bf16)
// ---------------------------------------------------------------------------
__global__ void gin_seg_init(float* __restrict__ fA,
                             const float* __restrict__ x,
                             const unsigned short* __restrict__ hB,
                             int Nn, int K, int KPAD, int useX)
{
    size_t i = (size_t)blockIdx.x * blockDim.x + threadIdx.x;
    size_t total = (size_t)Nn * (size_t)KPAD;
    if (i >= total) return;
    int node = (int)(i / (size_t)KPAD);
    int k    = (int)(i % (size_t)KPAD);
    float v = 0.f;
    if (k < K)
        v = useX ? x[(size_t)node * K + k] : bf2f(hB[(size_t)node * HDIM + k]);
    fA[i] = v;
}

// ---------------------------------------------------------------------------
// One block per edge: fA[dst] += feat(src). fp32 atomics (L2-resident rows).
// ---------------------------------------------------------------------------
__global__ void gin_edge_scatter(const int* __restrict__ ei,
                                 const float* __restrict__ x,
                                 const unsigned short* __restrict__ hB,
                                 float* __restrict__ fA,
                                 int Ee, int K, int KPAD, int useX)
{
    int e = blockIdx.x;
    int s = ei[e];
    int d = ei[(size_t)Ee + e];
    for (int k = threadIdx.x; k < K; k += blockDim.x) {
        float v = useX ? x[(size_t)s * K + k] : bf2f(hB[(size_t)s * HDIM + k]);
        atomicAdd(fA + (size_t)d * KPAD + k, v);
    }
}

__global__ void gin_to_bf16(const float* __restrict__ src,
                            unsigned short* __restrict__ dst, size_t n)
{
    size_t i = (size_t)blockIdx.x * blockDim.x + threadIdx.x;
    if (i < n) dst[i] = f2bf(src[i]);
}

// ---------------------------------------------------------------------------
// WMMA GEMM:  Out[rows x 128] = relu( (A[rows x KPAD] * Wt^T) * c1 + c0 )
// A, Wt bf16; acc f32 via v_wmma_f32_16x16x32_bf16.
// Block = 256 threads = 8 waves; wave w owns rows [blk*128 + 16w, +16),
// all 128 output columns (8 accumulators). Wt staged in LDS.
// B fragments for each K-step are batch-loaded before the WMMA chain so the
// ds_load_b128s pipeline instead of serializing on s_wait_dscnt 0.
// ---------------------------------------------------------------------------
template <int KPAD>
__global__ __launch_bounds__(256)
void gin_gemm_epilogue(const unsigned short* __restrict__ A,
                       const unsigned short* __restrict__ Wt,   // [128 x KPAD]
                       const float* __restrict__ c0,
                       const float* __restrict__ c1,
                       unsigned short* __restrict__ Out,        // [rows x 128]
                       int rows)
{
    extern __shared__ unsigned short sW[];                       // 128*KPAD bf16
    const int tid = threadIdx.x;

    const int nChunks = (HDIM * KPAD) / 8;                       // v4u chunks
    for (int i = tid; i < nChunks; i += 256)
        ((v4u*)sW)[i] = ((const v4u*)Wt)[i];
    __syncthreads();

    const int wave = tid >> 5;
    const int lane = tid & 31;
    const int lrow = lane & 15;
    const int hi   = (lane >> 4) << 3;                           // 0 or 8

    const int m0 = blockIdx.x * 128 + wave * 16;
    int arow = m0 + lrow;
    if (arow >= rows) arow = rows - 1;                           // clamp loads
    const unsigned short* Arow = A + (size_t)arow * KPAD;

    v8f acc[8];
#pragma unroll
    for (int nb = 0; nb < 8; ++nb)
        acc[nb] = (v8f){0.f, 0.f, 0.f, 0.f, 0.f, 0.f, 0.f, 0.f};

    for (int k0 = 0; k0 < KPAD; k0 += 32) {
        Frag af;
        af.u.lo = *(const v4u*)(Arow + k0 + hi);
        af.u.hi = *(const v4u*)(Arow + k0 + 16 + hi);

        Frag bfr[8];
#pragma unroll
        for (int nb = 0; nb < 8; ++nb) {
            const unsigned short* Wrow = sW + (size_t)(nb * 16 + lrow) * KPAD;
            bfr[nb].u.lo = *(const v4u*)(Wrow + k0 + hi);
            bfr[nb].u.hi = *(const v4u*)(Wrow + k0 + 16 + hi);
        }
#pragma unroll
        for (int nb = 0; nb < 8; ++nb)
            acc[nb] = __builtin_amdgcn_wmma_f32_16x16x32_bf16(
                false, af.v, false, bfr[nb].v, (short)0, acc[nb], false, false);
    }

    const int colbase = lane & 15;
    const int rowadd  = (lane >> 4) << 3;                        // 0 or 8
#pragma unroll
    for (int nb = 0; nb < 8; ++nb) {
        const int col = nb * 16 + colbase;
        const float s = c1[col];
        const float b = c0[col];
#pragma unroll
        for (int i = 0; i < 8; ++i) {
            int r = m0 + i + rowadd;
            if (r < rows) {
                float v = acc[nb][i] * s + b;
                v = v > 0.f ? v : 0.f;
                Out[(size_t)r * HDIM + col] = f2bf(v);
            }
        }
    }
}

// ---------------------------------------------------------------------------
// Global mean pool + final linear
// ---------------------------------------------------------------------------
__global__ void gin_pool_zero(float* __restrict__ sums, float* __restrict__ cnts, int Gg)
{
    int i = blockIdx.x * blockDim.x + threadIdx.x;
    if (i < Gg * HDIM) sums[i] = 0.f;
    if (i < Gg)        cnts[i] = 0.f;
}

__global__ void gin_pool_acc(const unsigned short* __restrict__ hB,
                             const int* __restrict__ batch,
                             float* __restrict__ sums, float* __restrict__ cnts, int Nn)
{
    int node = blockIdx.x;
    int k = threadIdx.x;                                          // blockDim = 128
    int g = batch[node];
    atomicAdd(sums + (size_t)g * HDIM + k, bf2f(hB[(size_t)node * HDIM + k]));
    if (k == 0) atomicAdd(cnts + g, 1.f);
}

__global__ void gin_pool_final(const float* __restrict__ sums,
                               const float* __restrict__ cnts,
                               const float* __restrict__ Wl,
                               const float* __restrict__ bl,
                               float* __restrict__ out, int Gg)
{
    int gwave = (int)(((size_t)blockIdx.x * blockDim.x + threadIdx.x) >> 5);
    int lane = threadIdx.x & 31;
    if (gwave >= Gg) return;
    float dot = 0.f;
    for (int k = lane; k < HDIM; k += 32)
        dot += sums[(size_t)gwave * HDIM + k] * Wl[k];
#pragma unroll
    for (int off = 16; off; off >>= 1)
        dot += __shfl_down(dot, off, 32);
    if (lane == 0) {
        float c = cnts[gwave];
        c = c > 1.f ? c : 1.f;
        out[gwave] = dot / c + bl[0];
    }
}

// ---------------------------------------------------------------------------
extern "C" void kernel_launch(void* const* d_in, const int* in_sizes, int n_in,
                              void* d_out, int out_size, void* d_ws, size_t ws_size,
                              hipStream_t stream)
{
    const float* x    = (const float*)d_in[0];
    const int* ei     = (const int*)d_in[1];
    const int* batch  = (const int*)d_in[2];

    const int Nn  = in_sizes[2];
    const int Fin = in_sizes[0] / Nn;          // 226
    const int Ee  = in_sizes[1] / 2;
    const int Gg  = out_size;                  // 512

    const float* P[32];
    for (int l = 0; l < 4; ++l)
        for (int j = 0; j < 8; ++j)
            P[l * 8 + j] = (const float*)d_in[3 + l * 8 + j];
    const float* Wl = (const float*)d_in[35];
    const float* bl = (const float*)d_in[36];

    // --- workspace carve (256-byte aligned) ---
    char* w = (char*)d_ws;
    auto alloc = [&](size_t bytes) -> char* {
        char* p = w;
        w += (bytes + 255) & ~(size_t)255;
        return p;
    };
    float*          fA  = (float*)alloc((size_t)Nn * 256 * sizeof(float));
    unsigned short* bA  = (unsigned short*)alloc((size_t)Nn * 256 * sizeof(unsigned short));
    unsigned short* hB  = (unsigned short*)alloc((size_t)Nn * HDIM * sizeof(unsigned short));
    unsigned short* WtA = (unsigned short*)alloc(HDIM * 256 * sizeof(unsigned short));
    unsigned short* WtB = (unsigned short*)alloc(HDIM * HDIM * sizeof(unsigned short));
    float* c0a = (float*)alloc(HDIM * sizeof(float));
    float* c1a = (float*)alloc(HDIM * sizeof(float));
    float* c0b = (float*)alloc(HDIM * sizeof(float));
    float* c1b = (float*)alloc(HDIM * sizeof(float));
    float* sums = (float*)alloc((size_t)Gg * HDIM * sizeof(float));
    float* cnts = (float*)alloc((size_t)Gg * sizeof(float));
    // mid-activation buffer aliases fA (fA is dead after bf16 conversion)
    unsigned short* mB = (unsigned short*)fA;

    for (int l = 0; l < 4; ++l) {
        const int din  = (l == 0) ? Fin : HDIM;
        const int KPAD = (l == 0) ? 256 : HDIM;
        const float *Wa = P[l*8+0], *ba = P[l*8+1], *g  = P[l*8+2], *be = P[l*8+3],
                    *rm = P[l*8+4], *rv = P[l*8+5], *Wb = P[l*8+6], *bb = P[l*8+7];

        int tw = HDIM * KPAD;
        gin_prep_weights<<<(tw + 255) / 256, 256, 0, stream>>>(
            Wa, din, KPAD, Wb, ba, g, be, rm, rv, bb, WtA, WtB, c0a, c1a, c0b, c1b);

        size_t tot = (size_t)Nn * (size_t)KPAD;
        gin_seg_init<<<(unsigned)((tot + 255) / 256), 256, 0, stream>>>(
            fA, x, hB, Nn, din, KPAD, l == 0 ? 1 : 0);

        gin_edge_scatter<<<Ee, 128, 0, stream>>>(ei, x, hB, fA, Ee, din, KPAD, l == 0 ? 1 : 0);

        gin_to_bf16<<<(unsigned)((tot + 255) / 256), 256, 0, stream>>>(fA, bA, tot);

        int blocks = (Nn + 127) / 128;
        if (KPAD == 256)
            gin_gemm_epilogue<256><<<blocks, 256, HDIM * 256 * 2, stream>>>(
                bA, WtA, c0a, c1a, mB, Nn);
        else
            gin_gemm_epilogue<128><<<blocks, 256, HDIM * 128 * 2, stream>>>(
                bA, WtA, c0a, c1a, mB, Nn);

        gin_gemm_epilogue<128><<<blocks, 256, HDIM * 128 * 2, stream>>>(
            mB, WtB, c0b, c1b, hB, Nn);
    }

    gin_pool_zero<<<(Gg * HDIM + 255) / 256, 256, 0, stream>>>(sums, cnts, Gg);
    gin_pool_acc<<<Nn, 128, 0, stream>>>(hB, batch, sums, cnts, Nn);
    gin_pool_final<<<(Gg * 32 + 255) / 256, 256, 0, stream>>>(
        sums, cnts, Wl, bl, (float*)d_out, Gg);
}
